// EncoderSelfAttention_65146063946247
// MI455X (gfx1250) — matrix operand order repaired and checked by reading, not compiled, not run on hardware
//
#include <hip/hip_runtime.h>
#include <hip/hip_bf16.h>
#include <math.h>
#include <stdint.h>

// Problem constants (from reference): B=2, C=512, H=W=64 -> N=4096
#define B_DIM 2
#define C_DIM 512
#define N_DIM 4096

typedef __attribute__((ext_vector_type(16))) __bf16        v16bf;
typedef __attribute__((ext_vector_type(8)))  float         v8f;
typedef __attribute__((ext_vector_type(4)))  unsigned int  uint4v;

union BfPack { uint4v u[2]; v16bf v; };

__device__ __forceinline__ unsigned short f2bf(float f) {
  union { float f; unsigned int u; } x; x.f = f;
  unsigned int r = x.u + 0x7FFFu + ((x.u >> 16) & 1u);   // round-to-nearest-even
  return (unsigned short)(r >> 16);
}

// ---- WMMA operand loaders (wave32 VGPR layouts per CDNA5 ISA 7.12.2) ----
// A matrix 16(M) x 32(K) bf16, source row-major with given row stride (halves).
// lanes 0-15: M=lane, halves {K0..7, K16..23}; lanes 16-31: M=lane-16, {K8..15, K24..31}
__device__ __forceinline__ v16bf load_A16x32(const unsigned short* base, int stride) {
  const int lane = threadIdx.x & 31;
  const int row  = lane & 15;
  const int hi   = lane >> 4;
  const unsigned short* p = base + (size_t)row * stride + hi * 8;
  BfPack r;
  r.u[0] = *(const uint4v*)(p);        // 8 halves
  r.u[1] = *(const uint4v*)(p + 16);   // 8 halves
  return r.v;
}

// B matrix 32(K) x 16(N) bf16, source stored K-contiguous per column:
// lane L holds column N=L&15, K {8h..8h+7, 16+8h..16+8h+7}, h=L>>4.
__device__ __forceinline__ v16bf load_B32x16(const unsigned short* base, int stride) {
  const int lane = threadIdx.x & 31;
  const int col  = lane & 15;
  const int hi   = lane >> 4;
  const unsigned short* p = base + (size_t)col * stride + hi * 8;
  BfPack r;
  r.u[0] = *(const uint4v*)(p);
  r.u[1] = *(const uint4v*)(p + 16);
  return r.v;
}

__device__ __forceinline__ v8f wmma_bf16(v16bf a, v16bf b, v8f c) {
  return __builtin_amdgcn_wmma_f32_16x16x32_bf16(false, a, false, b, (short)0, c, false, false);
}

// ------------------------------------------------------------------
// Kernel 1: x [B][C][N] f32  ->  xT [B][N][C] bf16 (LDS-tiled transpose)
// ------------------------------------------------------------------
__global__ __launch_bounds__(256) void transpose_x_kernel(const float* __restrict__ x,
                                                          unsigned short* __restrict__ xT) {
  __shared__ float tile[32][33];
  const int id = blockIdx.x;                 // B * (C/32) * (N/32) blocks
  const int b  = id / (16 * 128);
  const int r  = id % (16 * 128);
  const int ct = r / 128;
  const int nt = r % 128;
  const int tx = threadIdx.x, ty = threadIdx.y;
#pragma unroll
  for (int i = 0; i < 4; ++i) {
    int c = ct * 32 + ty + 8 * i;
    int n = nt * 32 + tx;
    tile[ty + 8 * i][tx] = x[((size_t)(b * C_DIM + c)) * N_DIM + n];
  }
  __syncthreads();
#pragma unroll
  for (int i = 0; i < 4; ++i) {
    int n = nt * 32 + ty + 8 * i;
    int c = ct * 32 + tx;
    xT[((size_t)(b * N_DIM + n)) * C_DIM + c] = f2bf(tile[tx][ty + 8 * i]);
  }
}

// ------------------------------------------------------------------
// Kernel 2: convert the four C x C f32 weight matrices to bf16 (contiguous dst)
// ------------------------------------------------------------------
__global__ __launch_bounds__(256) void cvt_w_kernel(const float* __restrict__ w0,
                                                    const float* __restrict__ w1,
                                                    const float* __restrict__ w2,
                                                    const float* __restrict__ w3,
                                                    unsigned short* __restrict__ dst) {
  const int idx = blockIdx.x * 256 + threadIdx.x;       // 4 * C*C total
  const int m = idx >> 18;                              // C*C = 2^18
  const int e = idx & ((1 << 18) - 1);
  const float* s = (m == 0) ? w0 : (m == 1) ? w1 : (m == 2) ? w2 : w3;
  dst[idx] = f2bf(s[e]);
}

// ------------------------------------------------------------------
// Kernel 3: q/k projections in transposed layout.
// outT[b][n][d] = alpha * ( sum_c xT[b][n][c] * W[d][c] + bias[d] )
// WMMA: A = xT rows n, K=c (contiguous);  B = W rows (row-major -> K-contiguous per d)
// D layout: M=n, Ncol=d -> contiguous bf16 stores along d.
// ------------------------------------------------------------------
__global__ __launch_bounds__(128) void proj_nt_kernel(const unsigned short* __restrict__ xT,
                                                      const unsigned short* __restrict__ W,
                                                      const float* __restrict__ bias,
                                                      unsigned short* __restrict__ outT,
                                                      float alpha) {
  const int gw   = (blockIdx.x * 128 + threadIdx.x) >> 5;  // global wave id
  const int lane = threadIdx.x & 31;
  const int hi = lane >> 4, lo = lane & 15;
  const int b  = gw / (256 * 8);                 // 256 n-tiles, 8 d-groups of 64
  const int r  = gw % (256 * 8);
  const int n0 = (r >> 3) << 4;
  const int d0 = (r & 7) << 6;

  const unsigned short* aBase = xT + ((size_t)(b * N_DIM + n0)) * C_DIM;
  const v8f zero = {0, 0, 0, 0, 0, 0, 0, 0};
  v8f acc[4] = {zero, zero, zero, zero};
  for (int c0 = 0; c0 < C_DIM; c0 += 32) {
    v16bf A = load_A16x32(aBase + c0, C_DIM);
#pragma unroll
    for (int j = 0; j < 4; ++j) {
      v16bf Bt = load_B32x16(W + (size_t)(d0 + 16 * j) * C_DIM + c0, C_DIM);
      acc[j] = wmma_bf16(A, Bt, acc[j]);
    }
  }
#pragma unroll
  for (int j = 0; j < 4; ++j) {
    const int d = d0 + 16 * j + lo;
    const float bv = bias[d];
#pragma unroll
    for (int i = 0; i < 8; ++i) {
      const int n = n0 + i + 8 * hi;
      outT[((size_t)(b * N_DIM + n)) * C_DIM + d] = f2bf((acc[j][i] + bv) * alpha);
    }
  }
}

// ------------------------------------------------------------------
// Kernel 4: [d][n]-layout projection (used for V and for the final O proj).
// out[b][d][n] = sum_c W[d][c] * T[b][n][c] + bias[d]
// WMMA: A = W rows d (K=c contiguous); B = T (K=c contiguous per column n).
// ------------------------------------------------------------------
__device__ __forceinline__ void store_out(float* p, float v) { *p = v; }
__device__ __forceinline__ void store_out(unsigned short* p, float v) { *p = f2bf(v); }

template <typename OutT>
__global__ __launch_bounds__(128) void proj_dn_kernel(const unsigned short* __restrict__ T,
                                                      const unsigned short* __restrict__ W,
                                                      const float* __restrict__ bias,
                                                      OutT* __restrict__ out) {
  const int gw   = (blockIdx.x * 128 + threadIdx.x) >> 5;
  const int lane = threadIdx.x & 31;
  const int hi = lane >> 4, lo = lane & 15;
  const int b  = gw / (32 * 64);                 // 32 d-tiles, 64 n-groups of 64
  const int r  = gw % (32 * 64);
  const int d0 = (r >> 6) << 4;
  const int n0 = (r & 63) << 6;

  const v8f zero = {0, 0, 0, 0, 0, 0, 0, 0};
  v8f acc[4] = {zero, zero, zero, zero};
  for (int c0 = 0; c0 < C_DIM; c0 += 32) {
    v16bf A = load_A16x32(W + (size_t)d0 * C_DIM + c0, C_DIM);
#pragma unroll
    for (int j = 0; j < 4; ++j) {
      v16bf Bt = load_B32x16(T + ((size_t)(b * N_DIM + n0 + 16 * j)) * C_DIM + c0, C_DIM);
      acc[j] = wmma_bf16(A, Bt, acc[j]);
    }
  }
#pragma unroll
  for (int i = 0; i < 8; ++i) {
    const int d = d0 + i + 8 * hi;
    const float bv = bias[d];
#pragma unroll
    for (int j = 0; j < 4; ++j) {
      const int n = n0 + 16 * j + lo;
      store_out(out + ((size_t)(b * C_DIM + d)) * N_DIM + n, acc[j][i] + bv);
    }
  }
}

// ------------------------------------------------------------------
// Kernel 5: flash attention. One workgroup (4 waves) per 16-query block.
// Scores contraction over C=512 is split across the 4 waves (128 each),
// reduced in LDS; online softmax; each wave owns a 128-wide slice of C
// for the P*V accumulation. Result written as oT[b][n][c] bf16.
// ------------------------------------------------------------------
__global__ __launch_bounds__(128) void attn_kernel(const unsigned short* __restrict__ qT,
                                                   const unsigned short* __restrict__ kT,
                                                   const unsigned short* __restrict__ vcn,
                                                   unsigned short* __restrict__ oT) {
  __shared__ float part[4][16][32];                        // per-wave partial scores
  __shared__ __align__(16) unsigned short probs[16][32];   // bf16 softmax probs
  __shared__ float rowM[16], rowL[16], rowScale[16];

  const int b  = blockIdx.x >> 8;            // 256 query tiles per batch
  const int n0 = (blockIdx.x & 255) << 4;
  const int tid  = threadIdx.x;
  const int wave = tid >> 5;
  const int lane = tid & 31;
  const int hi = lane >> 4, lo = lane & 15;
  const int cslice = wave * 128;             // this wave's channel slice

  if (tid < 16) { rowM[tid] = -3.0e38f; rowL[tid] = 0.0f; }

  const v8f zero = {0, 0, 0, 0, 0, 0, 0, 0};
  v8f acc[8];
#pragma unroll
  for (int t = 0; t < 8; ++t) acc[t] = zero;

  // Keep the 16 x 128 Q slab resident in VGPRs for the whole key sweep.
  const unsigned short* qbase = qT + ((size_t)(b * N_DIM + n0)) * C_DIM + cslice;
  v16bf Aq[4];
#pragma unroll
  for (int kc = 0; kc < 4; ++kc) Aq[kc] = load_A16x32(qbase + 32 * kc, C_DIM);

  __syncthreads();

  for (int m0 = 0; m0 < N_DIM; m0 += 32) {
    // ---- partial S = Q*K^T on this wave's 128-wide C slice ----
#pragma unroll
    for (int mt = 0; mt < 2; ++mt) {
      const unsigned short* kbase =
          kT + ((size_t)(b * N_DIM + m0 + 16 * mt)) * C_DIM + cslice;
      v8f s = zero;
#pragma unroll
      for (int kc = 0; kc < 4; ++kc) {
        v16bf Bk = load_B32x16(kbase + 32 * kc, C_DIM);
        s = wmma_bf16(Aq[kc], Bk, s);
      }
#pragma unroll
      for (int i = 0; i < 8; ++i)
        part[wave][i + 8 * hi][16 * mt + lo] = s[i];
    }
    if (m0 + 32 < N_DIM)   // hint next K tile into cache (global_prefetch_b8)
      __builtin_prefetch(kT + ((size_t)(b * N_DIM + m0 + 32)) * C_DIM + cslice + lane * 16, 0, 0);
    __syncthreads();

    // ---- online softmax: one thread per query row over the 32 new keys ----
    if (tid < 16) {
      const int rrow = tid;
      float sv[32];
      float mOld = rowM[rrow];
      float mNew = mOld;
#pragma unroll
      for (int j = 0; j < 32; ++j) {
        float t = part[0][rrow][j] + part[1][rrow][j] + part[2][rrow][j] + part[3][rrow][j];
        sv[j] = t;
        mNew = fmaxf(mNew, t);
      }
      const float sc = __expf(mOld - mNew);
      float lsum = 0.0f;
#pragma unroll
      for (int j = 0; j < 32; ++j) {
        float p = __expf(sv[j] - mNew);
        lsum += p;
        probs[rrow][j] = f2bf(p);
      }
      rowL[rrow] = rowL[rrow] * sc + lsum;
      rowM[rrow] = mNew;
      rowScale[rrow] = sc;
    }
    __syncthreads();

    // ---- rescale accumulators, then O += P * V on this wave's C slice ----
    float f[8];
#pragma unroll
    for (int i = 0; i < 8; ++i) f[i] = rowScale[i + 8 * hi];
    v16bf Ap = load_A16x32(&probs[0][0], 32);
#pragma unroll
    for (int t = 0; t < 8; ++t) {
#pragma unroll
      for (int i = 0; i < 8; ++i) acc[t][i] *= f[i];
      const unsigned short* vbase =
          vcn + ((size_t)(b * C_DIM + cslice + 16 * t)) * N_DIM + m0;
      v16bf Bv = load_B32x16(vbase, N_DIM);
      acc[t] = wmma_bf16(Ap, Bv, acc[t]);
    }
  }

  __syncthreads();
  float linv[8];
#pragma unroll
  for (int i = 0; i < 8; ++i) linv[i] = 1.0f / rowL[i + 8 * hi];
#pragma unroll
  for (int t = 0; t < 8; ++t) {
#pragma unroll
    for (int i = 0; i < 8; ++i) {
      const int n = n0 + i + 8 * hi;
      const int c = cslice + 16 * t + lo;
      oT[((size_t)(b * N_DIM + n)) * C_DIM + c] = f2bf(acc[t][i] * linv[i]);
    }
  }
}

// ------------------------------------------------------------------
extern "C" void kernel_launch(void* const* d_in, const int* in_sizes, int n_in,
                              void* d_out, int out_size, void* d_ws, size_t ws_size,
                              hipStream_t stream) {
  (void)in_sizes; (void)n_in; (void)out_size; (void)ws_size;

  const float* x  = (const float*)d_in[0];
  const float* wq = (const float*)d_in[1];
  const float* bq = (const float*)d_in[2];
  const float* wk = (const float*)d_in[3];
  const float* bk = (const float*)d_in[4];
  const float* wv = (const float*)d_in[5];
  const float* bv = (const float*)d_in[6];
  const float* wo = (const float*)d_in[7];
  const float* bo = (const float*)d_in[8];
  float* out = (float*)d_out;

  const size_t tsz = (size_t)B_DIM * N_DIM * C_DIM;   // elements per bf16 tensor
  unsigned short* xT  = (unsigned short*)d_ws;
  unsigned short* qT  = xT + tsz;
  unsigned short* kT  = qT + tsz;
  unsigned short* vcn = kT + tsz;
  unsigned short* oT  = vcn + tsz;
  unsigned short* wbf = oT + tsz;                     // 4 * C*C bf16 weights
  unsigned short* wqb = wbf;
  unsigned short* wkb = wbf + (size_t)C_DIM * C_DIM;
  unsigned short* wvb = wbf + 2 * (size_t)C_DIM * C_DIM;
  unsigned short* wob = wbf + 3 * (size_t)C_DIM * C_DIM;

  const float scale = 1.0f / sqrtf((float)C_DIM);     // folded into q

  // 1) x -> xT bf16
  transpose_x_kernel<<<B_DIM * 16 * 128, dim3(32, 8), 0, stream>>>(x, xT);
  // 2) weights -> bf16
  cvt_w_kernel<<<(4 * C_DIM * C_DIM) / 256, 256, 0, stream>>>(wq, wk, wv, wo, wbf);
  // 3) q (scale folded) and k projections in [N][C] layout
  proj_nt_kernel<<<1024, 128, 0, stream>>>(xT, wqb, bq, qT, scale);
  proj_nt_kernel<<<1024, 128, 0, stream>>>(xT, wkb, bk, kT, 1.0f);
  // 4) v projection in [C][N] layout
  proj_dn_kernel<unsigned short><<<1024, 128, 0, stream>>>(xT, wvb, bv, vcn);
  // 5) flash attention -> oT [N][C] bf16
  attn_kernel<<<B_DIM * (N_DIM / 16), 128, 0, stream>>>(qT, kT, vcn, oT);
  // 6) output projection -> d_out [B][C][N] f32
  proj_dn_kernel<float><<<1024, 128, 0, stream>>>(oT, wob, bo, out);
}
